// Net_17051020165258
// MI455X (gfx1250) — matrix operand order repaired and checked by reading, not compiled, not run on hardware
//
#include <hip/hip_runtime.h>

typedef __attribute__((ext_vector_type(16))) _Float16 v16h;
typedef __attribute__((ext_vector_type(8)))  _Float16 v8h;
typedef __attribute__((ext_vector_type(8)))  float    v8f;

constexpr int B_G   = 64;
constexpr int N_G   = 512;
constexpr int EPG   = 8192;
constexpr int E_TOT = B_G * EPG;        // 524288
constexpr int BN    = B_G * N_G;        // 32768
constexpr int K1V   = 410;              // ceil(0.8*512)
constexpr int K2V   = 328;              // ceil(0.8*410)
constexpr int NK    = B_G * K1V;        // 26240
constexpr int NK2   = B_G * K2V;        // 20992
constexpr unsigned ENC_NEG_INF = 0x007FFFFFu;   // enc(-inf)

#define LDS_FENCE() asm volatile("s_wait_dscnt 0" ::: "memory")

// ---------------- monotone float<->uint for atomic max ----------------
__device__ __forceinline__ unsigned enc_f32(float f) {
    unsigned u = __float_as_uint(f);
    return (u & 0x80000000u) ? ~u : (u | 0x80000000u);
}
__device__ __forceinline__ float dec_f32(unsigned k) {
    unsigned u = (k & 0x80000000u) ? (k & 0x7FFFFFFFu) : ~k;
    return __uint_as_float(u);
}
__device__ __forceinline__ float lrelu02(float x) { return x > 0.0f ? x : 0.2f * x; }
__device__ __forceinline__ float relu(float x)    { return x > 0.0f ? x : 0.0f; }

// paired f16 store (4B, aligned: even column)
__device__ __forceinline__ void store_h2pk(_Float16* dst, float a, float b) {
    union { _Float16 h[2]; unsigned u; } p;
    p.h[0] = (_Float16)a; p.h[1] = (_Float16)b;
    *reinterpret_cast<unsigned*>(dst) = p.u;
}

// ---------------- WMMA fragment loads (wave32, 16x16x32 f16) ----------------
// A 16x32 f16 (ISA 7.12.2): lane L row M=L%16; elements 0..7 are K=kbase+g*8..+7,
// elements 8..15 are K=kbase+16+g*8..+7  ->  two 16B LDS loads per lane.
__device__ __forceinline__ v16h load_a_packed(const _Float16* t, int ld, int kbase,
                                              int lane) {
    int m = lane & 15, g = (lane >> 4) & 1;
    const _Float16* p = t + m * ld + kbase + g * 8;
    v8h lo = *(const v8h*)p;
    v8h hi = *(const v8h*)(p + 16);
    return __builtin_shufflevector(lo, hi, 0, 1, 2, 3, 4, 5, 6, 7,
                                   8, 9, 10, 11, 12, 13, 14, 15);
}
// B fragments pre-packed: fragment fi, lane l -> 16 contiguous halves (32B)
__device__ __forceinline__ v16h load_b_packed(const _Float16* Wp, int fi, int lane) {
    return ((const v16h*)Wp)[fi * 32 + lane];
}
#define WMMA_F16(c, a, b) \
    __builtin_amdgcn_wmma_f32_16x16x32_f16(false, (a), false, (b), (short)0, (c), false, false)

// =====================================================================
// Pack weights [K x N] f32 -> f16 fragment layout, zero pad K -> Kpad.
// fragment fi = kc*(N/16)+nt; lane l; elem e (i=e>>1,h=e&1):
//   k = kc*32 + (e>>3)*16 + (l>>4)*8 + (e&7) ; n = nt*16 + (l&15)
// =====================================================================
__global__ void k_pack_b(const float* __restrict__ W, int K, int N, int Kpad,
                         _Float16* __restrict__ out) {
    int idx = blockIdx.x * blockDim.x + threadIdx.x;
    int total = (Kpad / 32) * (N / 16) * 512;
    if (idx >= total) return;
    int fi = idx >> 9;
    int l  = (idx >> 4) & 31;
    int e  = idx & 15;
    int ntiles = N / 16;
    int kc = fi / ntiles, nt = fi - kc * ntiles;
    int n = nt * 16 + (l & 15);
    int k = kc * 32 + ((e >> 3) << 4) + ((l >> 4) << 3) + (e & 7);
    out[idx] = (_Float16)((k < K) ? W[k * N + n] : 0.0f);
}

// =====================================================================
// Kernel 0: init accumulators
// =====================================================================
__global__ void k_init(float* __restrict__ agg, float* __restrict__ outg,
                       unsigned* __restrict__ menc, float* __restrict__ den) {
    int i = blockIdx.x * blockDim.x + threadIdx.x;
    if (i < NK * 256) outg[i] = 0.0f;
    if (i < BN * 32)  agg[i]  = 0.0f;
    if (i < NK * 4)  { menc[i] = ENC_NEG_INF; den[i] = 0.0f; }
}

// =====================================================================
// Kernel 1: edge encoder (2 WMMA GEMMs) fused with GINE message scatter
// =====================================================================
#define WPB1 4
__global__ void k_edge_encoder(const float* __restrict__ eattr,
                               const int* __restrict__ ei,
                               const float* __restrict__ x,
                               const _Float16* __restrict__ wp1, const float* __restrict__ b1,
                               const _Float16* __restrict__ wp2, const float* __restrict__ b2,
                               float* __restrict__ agg) {
    __shared__ alignas(16) _Float16 Ast[WPB1][16][136];
    __shared__ alignas(16) _Float16 Mid[WPB1][16][72];
    __shared__ float                Ea [WPB1][16][36];
    int wave = threadIdx.x >> 5, lane = threadIdx.x & 31;
    int tile = blockIdx.x * WPB1 + wave;
    int e0 = tile * 16;

    __builtin_prefetch(eattr + (size_t)(e0 + 16) * 100, 0, 1);

    // stage 16x100 f32 -> f16, zero pad cols 100..127
    for (int q = lane; q < 16 * 64; q += 32) {
        int r = q >> 6, c2 = (q & 63) * 2;
        float f0 = (c2     < 100) ? eattr[(e0 + r) * 100 + c2]     : 0.0f;
        float f1 = (c2 + 1 < 100) ? eattr[(e0 + r) * 100 + c2 + 1] : 0.0f;
        store_h2pk(&Ast[wave][r][c2], f0, f1);
    }
    LDS_FENCE();

    v16h af[4];
#pragma unroll
    for (int kc = 0; kc < 4; ++kc)
        af[kc] = load_a_packed(&Ast[wave][0][0], 136, kc * 32, lane);

    int nloc = lane & 15, g = (lane >> 4) & 1;
#pragma unroll
    for (int nt = 0; nt < 4; ++nt) {
        v8f c = {};
#pragma unroll
        for (int kc = 0; kc < 4; ++kc)
            c = WMMA_F16(c, af[kc], load_b_packed(wp1, kc * 4 + nt, lane));
        int n = nt * 16 + nloc;
        float bias = b1[n];
#pragma unroll
        for (int r = 0; r < 8; ++r)
            Mid[wave][r + 8 * g][n] = (_Float16)relu(c[r] + bias);
    }
    LDS_FENCE();

    v16h a2[2];
#pragma unroll
    for (int kc = 0; kc < 2; ++kc)
        a2[kc] = load_a_packed(&Mid[wave][0][0], 72, kc * 32, lane);
#pragma unroll
    for (int nt = 0; nt < 2; ++nt) {
        v8f c = {};
#pragma unroll
        for (int kc = 0; kc < 2; ++kc)
            c = WMMA_F16(c, a2[kc], load_b_packed(wp2, kc * 2 + nt, lane));
        int n = nt * 16 + nloc;
        float bias = b2[n];
#pragma unroll
        for (int r = 0; r < 8; ++r)
            Ea[wave][r + 8 * g][n] = relu(c[r] + bias);
    }
    LDS_FENCE();

    // msg = relu(x[src] + ea); agg[dst] += msg (32 features == 32 lanes)
    for (int e = 0; e < 16; ++e) {
        int gs = ei[e0 + e];
        int gd = ei[E_TOT + e0 + e];
        float v = relu(x[gs * 32 + lane] + Ea[wave][e][lane]);
        atomicAdd(&agg[gd * 32 + lane], v);
    }
}

// =====================================================================
// Kernel 2: h = x + agg; h2 = relu(relu(h@W1+b1)@W2+b2)   (WMMA)
// =====================================================================
#define WPB2 4
__global__ void k_node_mlp(const float* __restrict__ x, const float* __restrict__ agg,
                           const _Float16* __restrict__ wp1, const float* __restrict__ b1,
                           const _Float16* __restrict__ wp2, const float* __restrict__ b2,
                           float* __restrict__ h2) {
    __shared__ alignas(16) _Float16 Hs [WPB2][16][40];
    __shared__ alignas(16) _Float16 Mid[WPB2][16][136];
    int wave = threadIdx.x >> 5, lane = threadIdx.x & 31;
    int tile = blockIdx.x * WPB2 + wave;
    int n0 = tile * 16;

    for (int q = lane; q < 16 * 16; q += 32) {
        int r = q >> 4, c2 = (q & 15) * 2;
        int base = ((n0 + r) * 32 + c2) >> 1;
        float2 xv = ((const float2*)x)[base];
        float2 av = ((const float2*)agg)[base];
        store_h2pk(&Hs[wave][r][c2], xv.x + av.x, xv.y + av.y);
    }
    LDS_FENCE();

    int nloc = lane & 15, g = (lane >> 4) & 1;
    v16h a = load_a_packed(&Hs[wave][0][0], 40, 0, lane);
#pragma unroll
    for (int nt = 0; nt < 8; ++nt) {
        v8f c = {};
        c = WMMA_F16(c, a, load_b_packed(wp1, nt, lane));
        int n = nt * 16 + nloc;
        float bias = b1[n];
#pragma unroll
        for (int r = 0; r < 8; ++r)
            Mid[wave][r + 8 * g][n] = (_Float16)relu(c[r] + bias);
    }
    LDS_FENCE();

    v16h a2[4];
#pragma unroll
    for (int kc = 0; kc < 4; ++kc)
        a2[kc] = load_a_packed(&Mid[wave][0][0], 136, kc * 32, lane);
#pragma unroll
    for (int nt = 0; nt < 16; ++nt) {
        v8f c = {};
#pragma unroll
        for (int kc = 0; kc < 4; ++kc)
            c = WMMA_F16(c, a2[kc], load_b_packed(wp2, kc * 16 + nt, lane));
        int n = nt * 16 + nloc;
        float bias = b2[n];
#pragma unroll
        for (int r = 0; r < 8; ++r)
            h2[(n0 + r + 8 * g) * 256 + n] = relu(c[r] + bias);
    }
}

// =====================================================================
// per-row score s = tanh(dot(row, w)/||w||)   (one wave / row)
// =====================================================================
__global__ void k_score(const float* __restrict__ feat, const float* __restrict__ w,
                        float* __restrict__ s, int nrows) {
    int row = (blockIdx.x * blockDim.x + threadIdx.x) >> 5;
    int lane = threadIdx.x & 31;
    if (row >= nrows) return;
    float nrm = 0.0f, dot = 0.0f;
    for (int j = lane; j < 256; j += 32) {
        float wv = w[j];
        nrm += wv * wv;
        dot += feat[row * 256 + j] * wv;
    }
#pragma unroll
    for (int off = 16; off > 0; off >>= 1) {
        nrm += __shfl_xor(nrm, off, 32);
        dot += __shfl_xor(dot, off, 32);
    }
    if (lane == 0) s[row] = tanhf(dot / sqrtf(nrm));
}

// =====================================================================
// exact top-K by rank counting (ncount <= 512), ties like lax.top_k
// =====================================================================
__global__ void k_topk(const float* __restrict__ sc, int ncount, int K,
                       int* __restrict__ sel, float* __restrict__ selv,
                       int* __restrict__ newid) {
    __shared__ float sb[512];
    int b = blockIdx.x, i = threadIdx.x;
    if (i < ncount) sb[i] = sc[b * ncount + i];
    __syncthreads();
    if (i < ncount) {
        float vi = sb[i];
        int rank = 0;
        for (int j = 0; j < ncount; ++j) {
            float vj = sb[j];
            rank += (vj > vi) || (vj == vi && j < i);
        }
        if (rank < K) {
            sel[b * K + rank]  = b * ncount + i;
            selv[b * K + rank] = vi;
        }
        if (newid) newid[b * ncount + i] = (rank < K) ? (b * K + rank) : -1;
    }
}

// =====================================================================
// Kernel 4: hp = h2[gperm]*topv ; xl = hp @ gat_w ; attention dots (WMMA)
// =====================================================================
#define WPB4 4
__global__ void k_gat_mm(const float* __restrict__ h2, const int* __restrict__ gperm,
                         const float* __restrict__ topv, const _Float16* __restrict__ gwp,
                         const float* __restrict__ att_s, const float* __restrict__ att_d,
                         float* __restrict__ xl, float* __restrict__ a_src,
                         float* __restrict__ a_dst) {
    __shared__ alignas(16) _Float16 Hp[WPB4][16][264];
    __shared__ int      Nd[WPB4][16];
    __shared__ float    Sv[WPB4][16];
    __shared__ float    As[WPB4][16][4];
    __shared__ float    Ad[WPB4][16][4];
    int wave = threadIdx.x >> 5, lane = threadIdx.x & 31;
    int tile = blockIdx.x * WPB4 + wave;
    int row0 = tile * 16;

    if (lane < 16) {
        Nd[wave][lane] = gperm[row0 + lane];
        Sv[wave][lane] = topv[row0 + lane];
#pragma unroll
        for (int h = 0; h < 4; ++h) { As[wave][lane][h] = 0.0f; Ad[wave][lane][h] = 0.0f; }
    }
    LDS_FENCE();
    for (int q = lane; q < 16 * 128; q += 32) {
        int r = q >> 7, j = q & 127;
        float sc = Sv[wave][r];
        float2 hv = ((const float2*)(h2 + (size_t)Nd[wave][r] * 256))[j];
        store_h2pk(&Hp[wave][r][2 * j], hv.x * sc, hv.y * sc);
    }
    LDS_FENCE();

    v16h af[8];
#pragma unroll
    for (int kc = 0; kc < 8; ++kc)
        af[kc] = load_a_packed(&Hp[wave][0][0], 264, kc * 32, lane);

    int nloc = lane & 15, g = (lane >> 4) & 1;
#pragma unroll
    for (int nt = 0; nt < 16; ++nt) {
        v8f c = {};
#pragma unroll
        for (int kc = 0; kc < 8; ++kc)
            c = WMMA_F16(c, af[kc], load_b_packed(gwp, kc * 16 + nt, lane));
        int n = nt * 16 + nloc;
        int head = n >> 6, nn = n & 63;
        float ws = att_s[head * 64 + nn], wd = att_d[head * 64 + nn];
#pragma unroll
        for (int r = 0; r < 8; ++r) {
            int m = r + 8 * g;
            float v = c[r];
            xl[(row0 + m) * 256 + n] = v;
            atomicAdd(&As[wave][m][head], v * ws);
            atomicAdd(&Ad[wave][m][head], v * wd);
        }
    }
    LDS_FENCE();
    if (lane < 16) {
#pragma unroll
        for (int h = 0; h < 4; ++h) {
            a_src[(row0 + lane) * 4 + h] = As[wave][lane][h];
            a_dst[(row0 + lane) * 4 + h] = Ad[wave][lane][h];
        }
    }
}

// =====================================================================
// GAT softmax: segment max / den / accumulate
// edges [0,E) original (validity via newid>=0), [E,E+NK) self loops
// =====================================================================
__global__ void k_att_max(const int* __restrict__ ei, const int* __restrict__ newid,
                          const float* __restrict__ as, const float* __restrict__ ad,
                          unsigned* __restrict__ menc) {
    int e = blockIdx.x * blockDim.x + threadIdx.x;
    if (e >= E_TOT + NK) return;
    int s1, d1;
    if (e < E_TOT) {
        s1 = newid[ei[e]];
        d1 = newid[ei[E_TOT + e]];
        if (s1 < 0 || d1 < 0) return;
    } else {
        s1 = d1 = e - E_TOT;
    }
#pragma unroll
    for (int h = 0; h < 4; ++h) {
        float l = lrelu02(as[s1 * 4 + h] + ad[d1 * 4 + h]);
        atomicMax(&menc[d1 * 4 + h], enc_f32(l));
    }
}

__global__ void k_att_den(const int* __restrict__ ei, const int* __restrict__ newid,
                          const float* __restrict__ as, const float* __restrict__ ad,
                          const unsigned* __restrict__ menc, float* __restrict__ den) {
    int e = blockIdx.x * blockDim.x + threadIdx.x;
    if (e >= E_TOT + NK) return;
    int s1, d1;
    if (e < E_TOT) {
        s1 = newid[ei[e]];
        d1 = newid[ei[E_TOT + e]];
        if (s1 < 0 || d1 < 0) return;
    } else {
        s1 = d1 = e - E_TOT;
    }
#pragma unroll
    for (int h = 0; h < 4; ++h) {
        float l = lrelu02(as[s1 * 4 + h] + ad[d1 * 4 + h]);
        atomicAdd(&den[d1 * 4 + h], expf(l - dec_f32(menc[d1 * 4 + h])));
    }
}

__global__ void k_att_acc(const int* __restrict__ ei, const int* __restrict__ newid,
                          const float* __restrict__ as, const float* __restrict__ ad,
                          const unsigned* __restrict__ menc, const float* __restrict__ den,
                          const float* __restrict__ xl, float* __restrict__ outg) {
    int e = blockIdx.x, f = threadIdx.x;
    int s1, d1;
    if (e < E_TOT) {
        s1 = newid[ei[e]];
        d1 = newid[ei[E_TOT + e]];
        if (s1 < 0 || d1 < 0) return;
    } else {
        s1 = d1 = e - E_TOT;
    }
    int h = f >> 6;
    float l = lrelu02(as[s1 * 4 + h] + ad[d1 * 4 + h]);
    float alpha = expf(l - dec_f32(menc[d1 * 4 + h])) / den[d1 * 4 + h];
    atomicAdd(&outg[d1 * 256 + f], alpha * xl[s1 * 256 + f]);
}

// g = relu(out + gat_b) in place ; s2 = tanh(dot(g,pool2_w)/||pool2_w||)
__global__ void k_bias_relu_score(float* __restrict__ outg, const float* __restrict__ gb,
                                  const float* __restrict__ pw2, float* __restrict__ s2) {
    int row = (blockIdx.x * blockDim.x + threadIdx.x) >> 5;
    int lane = threadIdx.x & 31;
    if (row >= NK) return;
    float nrm = 0.0f, dot = 0.0f;
    for (int j = lane; j < 256; j += 32) {
        float v = relu(outg[row * 256 + j] + gb[j]);
        outg[row * 256 + j] = v;
        float wv = pw2[j];
        nrm += wv * wv;
        dot += v * wv;
    }
#pragma unroll
    for (int off = 16; off > 0; off >>= 1) {
        nrm += __shfl_xor(nrm, off, 32);
        dot += __shfl_xor(dot, off, 32);
    }
    if (lane == 0) s2[row] = tanhf(dot / sqrtf(nrm));
}

// =====================================================================
// readouts (max+mean), z = x1+x2, 3-layer head, log_softmax
// =====================================================================
__global__ void k_head(const float* __restrict__ h2, const int* __restrict__ gperm,
                       const float* __restrict__ topv, const float* __restrict__ g,
                       const int* __restrict__ sel2, const float* __restrict__ topv2,
                       const float* __restrict__ w1, const float* __restrict__ b1,
                       const float* __restrict__ w2, const float* __restrict__ b2,
                       const float* __restrict__ w3, const float* __restrict__ b3,
                       float* __restrict__ out) {
    __shared__ int   nd[K1V];
    __shared__ float sc[K1V];
    __shared__ int   rw[K2V];
    __shared__ float sc2[K2V];
    __shared__ float z[512];
    __shared__ float a1[128];
    __shared__ float a2v[64];
    int b = blockIdx.x, t = threadIdx.x;
    for (int i = t; i < K1V; i += 256) { nd[i] = gperm[b * K1V + i]; sc[i] = topv[b * K1V + i]; }
    for (int i = t; i < K2V; i += 256) { rw[i] = sel2[b * K2V + i]; sc2[i] = topv2[b * K2V + i]; }
    __syncthreads();

    float hmax = -INFINITY, hsum = 0.0f;
    for (int r = 0; r < K1V; ++r) {
        float v = h2[nd[r] * 256 + t] * sc[r];
        hmax = fmaxf(hmax, v);
        hsum += v;
    }
    float gmax = -INFINITY, gsum = 0.0f;
    for (int r = 0; r < K2V; ++r) {
        float v = g[rw[r] * 256 + t] * sc2[r];
        gmax = fmaxf(gmax, v);
        gsum += v;
    }
    z[t]       = hmax + gmax;
    z[256 + t] = hsum / (float)K1V + gsum / (float)K2V;
    __syncthreads();

    if (t < 128) {
        float acc = b1[t];
        for (int i = 0; i < 512; ++i) acc += z[i] * w1[i * 128 + t];
        a1[t] = relu(acc);
    }
    __syncthreads();
    if (t < 64) {
        float acc = b2[t];
        for (int i = 0; i < 128; ++i) acc += a1[i] * w2[i * 64 + t];
        a2v[t] = relu(acc);
    }
    __syncthreads();
    if (t == 0) {
        float l0 = b3[0], l1 = b3[1];
        for (int i = 0; i < 64; ++i) { l0 += a2v[i] * w3[i * 2]; l1 += a2v[i] * w3[i * 2 + 1]; }
        float mm = fmaxf(l0, l1);
        float lse = mm + logf(expf(l0 - mm) + expf(l1 - mm));
        out[b * 2 + 0] = l0 - lse;
        out[b * 2 + 1] = l1 - lse;
    }
}

// =====================================================================
extern "C" void kernel_launch(void* const* d_in, const int* in_sizes, int n_in,
                              void* d_out, int out_size, void* d_ws, size_t ws_size,
                              hipStream_t stream) {
    const float* x       = (const float*)d_in[0];
    const int*   ei      = (const int*)  d_in[1];
    const float* eattr   = (const float*)d_in[3];
    const float* dec1_w  = (const float*)d_in[4];
    const float* dec1_b  = (const float*)d_in[5];
    const float* dec2_w  = (const float*)d_in[6];
    const float* dec2_b  = (const float*)d_in[7];
    const float* mlp1_w  = (const float*)d_in[8];
    const float* mlp1_b  = (const float*)d_in[9];
    const float* mlp2_w  = (const float*)d_in[10];
    const float* mlp2_b  = (const float*)d_in[11];
    const float* pool1_w = (const float*)d_in[12];
    const float* gat_w   = (const float*)d_in[13];
    const float* att_s   = (const float*)d_in[14];
    const float* att_d   = (const float*)d_in[15];
    const float* gat_b   = (const float*)d_in[16];
    const float* pool2_w = (const float*)d_in[17];
    const float* lin1_w  = (const float*)d_in[18];
    const float* lin1_b  = (const float*)d_in[19];
    const float* lin2_w  = (const float*)d_in[20];
    const float* lin2_b  = (const float*)d_in[21];
    const float* lin3_w  = (const float*)d_in[22];
    const float* lin3_b  = (const float*)d_in[23];
    float* out = (float*)d_out;

    // ---- workspace carve-up ----
    char* ws = (char*)d_ws;
    size_t off = 0;
    auto carve = [&](size_t bytes) -> char* {
        char* p = ws + off;
        off = (off + bytes + 255) & ~(size_t)255;
        return p;
    };
    float*    agg     = (float*)   carve((size_t)BN * 32 * 4);
    float*    h2      = (float*)   carve((size_t)BN * 256 * 4);
    float*    s1      = (float*)   carve((size_t)BN * 4);
    int*      newid   = (int*)     carve((size_t)BN * 4);
    int*      gperm   = (int*)     carve((size_t)NK * 4);
    float*    topv    = (float*)   carve((size_t)NK * 4);
    float*    xl      = (float*)   carve((size_t)NK * 256 * 4);
    float*    a_src   = (float*)   carve((size_t)NK * 4 * 4);
    float*    a_dst   = (float*)   carve((size_t)NK * 4 * 4);
    unsigned* menc    = (unsigned*)carve((size_t)NK * 4 * 4);
    float*    den     = (float*)   carve((size_t)NK * 4 * 4);
    float*    outg    = (float*)   carve((size_t)NK * 256 * 4);
    float*    s2      = (float*)   carve((size_t)NK * 4);
    int*      sel2    = (int*)     carve((size_t)NK2 * 4);
    float*    topv2   = (float*)   carve((size_t)NK2 * 4);
    // packed f16 weights (fragment layout)
    _Float16* wp_dec1 = (_Float16*)carve((size_t)(128/32)*(64/16)  * 512 * 2);
    _Float16* wp_dec2 = (_Float16*)carve((size_t)(64/32) *(32/16)  * 512 * 2);
    _Float16* wp_mlp1 = (_Float16*)carve((size_t)(32/32) *(128/16) * 512 * 2);
    _Float16* wp_mlp2 = (_Float16*)carve((size_t)(128/32)*(256/16) * 512 * 2);
    _Float16* wp_gat  = (_Float16*)carve((size_t)(256/32)*(256/16) * 512 * 2);

    // weight packing
    auto pack = [&](const float* W, int K, int N, int Kpad, _Float16* o) {
        int total = (Kpad / 32) * (N / 16) * 512;
        k_pack_b<<<(total + 255) / 256, 256, 0, stream>>>(W, K, N, Kpad, o);
    };
    pack(dec1_w, 100,  64, 128, wp_dec1);
    pack(dec2_w,  64,  32,  64, wp_dec2);
    pack(mlp1_w,  32, 128,  32, wp_mlp1);
    pack(mlp2_w, 128, 256, 128, wp_mlp2);
    pack(gat_w,  256, 256, 256, wp_gat);

    // 0) init accumulators
    k_init<<<(NK * 256 + 255) / 256, 256, 0, stream>>>(agg, outg, menc, den);
    // 1) edge encoder + message scatter
    k_edge_encoder<<<(E_TOT / 16) / WPB1, 32 * WPB1, 0, stream>>>(
        eattr, ei, x, wp_dec1, dec1_b, wp_dec2, dec2_b, agg);
    // 2) GINE node MLP
    k_node_mlp<<<(BN / 16) / WPB2, 32 * WPB2, 0, stream>>>(
        x, agg, wp_mlp1, mlp1_b, wp_mlp2, mlp2_b, h2);
    // 2b) pooling-1 scores
    k_score<<<(BN * 32) / 256, 256, 0, stream>>>(h2, pool1_w, s1, BN);
    // 3) top-K1 selection + relabeling
    k_topk<<<B_G, 512, 0, stream>>>(s1, N_G, K1V, gperm, topv, newid);
    // 4) GAT projection + attention dots
    k_gat_mm<<<(NK / 16) / WPB4, 32 * WPB4, 0, stream>>>(
        h2, gperm, topv, wp_gat, att_s, att_d, xl, a_src, a_dst);
    // 5) GAT edge softmax (max, den, accumulate)
    {
        int ne = E_TOT + NK;
        k_att_max<<<(ne + 255) / 256, 256, 0, stream>>>(ei, newid, a_src, a_dst, menc);
        k_att_den<<<(ne + 255) / 256, 256, 0, stream>>>(ei, newid, a_src, a_dst, menc, den);
        k_att_acc<<<ne, 256, 0, stream>>>(ei, newid, a_src, a_dst, menc, den, xl, outg);
    }
    // 5d) bias+relu and pooling-2 scores
    k_bias_relu_score<<<(NK * 32 + 255) / 256, 256, 0, stream>>>(outg, gat_b, pool2_w, s2);
    // 6) top-K2 selection
    k_topk<<<B_G, 512, 0, stream>>>(s2, K1V, K2V, sel2, topv2, nullptr);
    // 7) readouts + MLP head + log_softmax
    k_head<<<B_G, 256, 0, stream>>>(h2, gperm, topv, outg, sel2, topv2,
                                    lin1_w, lin1_b, lin2_w, lin2_b, lin3_w, lin3_b, out);
}
